// GPT_Attention_87093346828866
// MI455X (gfx1250) — matrix operand order repaired and checked by reading, not compiled, not run on hardware
//
#include <hip/hip_runtime.h>

// ---------------------------------------------------------------------------
// Types
// ---------------------------------------------------------------------------
typedef __attribute__((ext_vector_type(16))) __bf16 v16bf;
typedef __attribute__((ext_vector_type(8)))  __bf16 v8bf;
typedef __attribute__((ext_vector_type(8)))  float  v8f;

// Problem constants (match reference)
#define BB   4
#define SS   2048
#define IXX  1024
#define OXX  1024
#define HH   16
#define DHH  64
#define MROWS (BB * SS)          // 8192
#define QKVW  (3 * OXX)          // 3072

// ---------------------------------------------------------------------------
// Helpers
// ---------------------------------------------------------------------------
__device__ __forceinline__ v16bf load_frag16(const __bf16* p0, const __bf16* p1) {
    v8bf lo = *(const v8bf*)p0;
    v8bf hi = *(const v8bf*)p1;
    v16bf r;
#pragma unroll
    for (int i = 0; i < 8; ++i) { r[i] = lo[i]; r[8 + i] = hi[i]; }
    return r;
}

__device__ __forceinline__ v8f wmma_bf16(v16bf a, v16bf b, v8f c) {
    return __builtin_amdgcn_wmma_f32_16x16x32_bf16(false, a, false, b,
                                                   (short)0, c, false, false);
}

// Async global->LDS copy, 16 bytes per lane (tracked by ASYNCcnt).
__device__ __forceinline__ void async_copy_b128(__bf16* lds_dst, const __bf16* gsrc) {
    unsigned l = (unsigned)(uintptr_t)lds_dst;  // LDS offset = low 32 bits
    asm volatile("global_load_async_to_lds_b128 %0, %1, off"
                 :: "v"(l), "v"(gsrc) : "memory");
}
__device__ __forceinline__ void wait_async0() {
    asm volatile("s_wait_asynccnt 0" ::: "memory");
}

// xor-shuffle within 32 lanes via immediate ds_swizzle (group-of-32 mode:
// offset = {xor[4:0], or[4:0]=0, and[4:0]=0x1f}); masks < 16 stay in-half.
template <int MASK>
__device__ __forceinline__ float xor_shuffle(float x) {
    constexpr int imm = (MASK << 10) | 0x1f;
    return __int_as_float(__builtin_amdgcn_ds_swizzle(__float_as_int(x), imm));
}

// ---------------------------------------------------------------------------
// f32 -> bf16 conversion (n must be a multiple of 4)
// ---------------------------------------------------------------------------
__global__ void cvt_f32_bf16(const float* __restrict__ in,
                             __bf16* __restrict__ out, int n) {
    int i = (blockIdx.x * blockDim.x + threadIdx.x) * 4;
    if (i + 3 < n) {
        float4 f = *(const float4*)(in + i);
        out[i + 0] = (__bf16)f.x;
        out[i + 1] = (__bf16)f.y;
        out[i + 2] = (__bf16)f.z;
        out[i + 3] = (__bf16)f.w;
    }
}

// ---------------------------------------------------------------------------
// bf16 GEMM:  C[M,N] = A[M,K] @ Bm[K,N] + bias[N]
// block tile 128x128, 8 waves (4m x 2n), wave tile 32x64, K-step 32.
// Double-buffered LDS: A tile staged with async global->LDS DMA overlapped
// with WMMA compute on the other buffer; B tile transposed manually.
// ---------------------------------------------------------------------------
template <bool OUT_BF16>
__global__ __launch_bounds__(256)
void gemm_bf16_kernel(const __bf16* __restrict__ A,
                      const __bf16* __restrict__ Bm,
                      const float* __restrict__ bias,
                      void* __restrict__ Cout,
                      int M, int N, int K) {
    constexpr int LDA = 40;                          // padded row stride (halfs)
    __shared__ __align__(16) __bf16 aT[2][128 * LDA];   // [m][k], row major
    __shared__ __align__(16) __bf16 bT[2][128 * LDA];   // [n][k], B transposed

    const int tid  = threadIdx.x;
    const int wave = tid >> 5;
    const int lane = tid & 31;
    const int half = lane >> 4;
    const int l15  = lane & 15;
    const int wm   = (wave & 3) * 32;                // wave m-offset in block
    const int wn   = (wave >> 2) * 64;               // wave n-offset in block
    const int blockM = blockIdx.y * 128;
    const int blockN = blockIdx.x * 128;

    auto stage = [&](int buf, int kb) {
        // A tile via async DMA: rows blockM..+127, cols kb..+31
        {
            int row = tid >> 1;                      // 2 threads per row
            int ch  = (tid & 1) * 16;                // 16 halfs each
            const __bf16* src = A + (size_t)(blockM + row) * K + kb + ch;
            __bf16* dst = &aT[buf][row * LDA + ch];
            async_copy_b128(dst,     src);
            async_copy_b128(dst + 8, src + 8);
        }
        // B tile transposed: bT[n][k]
        {
            int k  = tid >> 3;                       // 8 threads per B row
            int ns = (tid & 7) * 16;
            const __bf16* src = Bm + (size_t)(kb + k) * N + blockN + ns;
            v8bf x0 = *(const v8bf*)src;
            v8bf x1 = *(const v8bf*)(src + 8);
#pragma unroll
            for (int i = 0; i < 8; ++i) bT[buf][(ns + i) * LDA + k]     = x0[i];
#pragma unroll
            for (int i = 0; i < 8; ++i) bT[buf][(ns + 8 + i) * LDA + k] = x1[i];
        }
    };

    v8f acc[2][4];
    const v8f zero8 = {0.f, 0.f, 0.f, 0.f, 0.f, 0.f, 0.f, 0.f};
#pragma unroll
    for (int mi = 0; mi < 2; ++mi)
#pragma unroll
        for (int ni = 0; ni < 4; ++ni) acc[mi][ni] = zero8;

    stage(0, 0);
    wait_async0();
    __syncthreads();

    for (int kb = 0; kb < K; kb += 32) {
        const int buf = (kb >> 5) & 1;
        if (kb + 32 < K) stage(buf ^ 1, kb + 32);    // DMA next tile in flight

        // ---- fragments from current buffer ----
        v16bf afr[2], bfr[4];
#pragma unroll
        for (int mi = 0; mi < 2; ++mi) {
            int row = wm + mi * 16 + l15;
            int k0  = half * 8;                      // A: half0 K{0-7,16-23}, half1 K{8-15,24-31}
            afr[mi] = load_frag16(&aT[buf][row * LDA + k0],
                                  &aT[buf][row * LDA + k0 + 16]);
        }
#pragma unroll
        for (int ni = 0; ni < 4; ++ni) {
            int col = wn + ni * 16 + l15;
            int k0  = half * 8;                      // B: half0 K{0-7,16-23}, half1 K{8-15,24-31}
            bfr[ni] = load_frag16(&bT[buf][col * LDA + k0],
                                  &bT[buf][col * LDA + k0 + 16]);
        }
#pragma unroll
        for (int mi = 0; mi < 2; ++mi)
#pragma unroll
            for (int ni = 0; ni < 4; ++ni)
                acc[mi][ni] = wmma_bf16(afr[mi], bfr[ni], acc[mi][ni]);

        wait_async0();
        __syncthreads();
    }

    // ---- epilogue: bias + store ----
#pragma unroll
    for (int mi = 0; mi < 2; ++mi)
#pragma unroll
        for (int ni = 0; ni < 4; ++ni) {
            int row0 = blockM + wm + mi * 16 + half * 8;
            int col  = blockN + wn + ni * 16 + l15;
            float bv = bias ? bias[col] : 0.f;
#pragma unroll
            for (int r = 0; r < 8; ++r) {
                float v = acc[mi][ni][r] + bv;
                if (OUT_BF16)
                    ((__bf16*)Cout)[(size_t)(row0 + r) * N + col] = (__bf16)v;
                else
                    ((float*)Cout)[(size_t)(row0 + r) * N + col] = v;
            }
        }
}

// ---------------------------------------------------------------------------
// Flash attention. Reference computes scores = K @ Q^T, softmax over t,
// out = attn @ V  =>  query-role = k-chunk (offset 0), key-role = q-chunk
// (offset OX), value = v-chunk (offset 2*OX) of the fused qkv buffer.
// Grid: (S/128, B*H). 8 waves, each owns 16 query rows; 64-key K/V chunks
// double-buffered in LDS (K via async DMA overlapped with compute).
// ---------------------------------------------------------------------------
__global__ __launch_bounds__(256)
void attn_kernel(const __bf16* __restrict__ qkvh, __bf16* __restrict__ attnh) {
    constexpr int LKV = 72;                             // padded stride (halfs)
    __shared__ __align__(16) __bf16 kt[2][64 * LKV];    // key tile   [t][d]
    __shared__ __align__(16) __bf16 vt[2][64 * LKV];    // value tile [d][t]
    __shared__ __align__(16) __bf16 pt[8][16 * LKV];    // per-wave P tile [row][t]

    const int tid  = threadIdx.x;
    const int wave = tid >> 5;
    const int lane = tid & 31;
    const int half = lane >> 4;
    const int l15  = lane & 15;
    const int bh = blockIdx.y;
    const int b  = bh / HH;
    const int h  = bh % HH;
    const int s0 = blockIdx.x * 128 + wave * 16;

    const __bf16* qbase = qkvh + (size_t)b * SS * QKVW + (size_t)h * DHH; // k-chunk
    const __bf16* kbase = qbase + OXX;                                    // q-chunk
    const __bf16* vbase = qbase + 2 * OXX;                                // v-chunk

    auto stageKV = [&](int buf, int t0) {
        int t  = tid >> 2;
        int ds = (tid & 3) * 16;
        const __bf16* ks = kbase + (size_t)(t0 + t) * QKVW + ds;
        __bf16* kd = &kt[buf][t * LKV + ds];
        async_copy_b128(kd,     ks);
        async_copy_b128(kd + 8, ks + 8);
        const __bf16* vs = vbase + (size_t)(t0 + t) * QKVW + ds;
        v8bf y0 = *(const v8bf*)vs;
        v8bf y1 = *(const v8bf*)(vs + 8);
#pragma unroll
        for (int i = 0; i < 8; ++i) vt[buf][(ds + i) * LKV + t]     = y0[i];
#pragma unroll
        for (int i = 0; i < 8; ++i) vt[buf][(ds + 8 + i) * LKV + t] = y1[i];
    };

    // Q fragments for this wave's 16 rows (d = 0..63 -> two 16x32 A frags)
    v16bf qa[2];
    {
        const __bf16* qp = qbase + (size_t)(s0 + l15) * QKVW;
        int k0 = half * 8;
        qa[0] = load_frag16(qp + k0,      qp + k0 + 16);
        qa[1] = load_frag16(qp + 32 + k0, qp + 32 + k0 + 16);
    }

    const v8f zero8 = {0.f, 0.f, 0.f, 0.f, 0.f, 0.f, 0.f, 0.f};
    v8f o[4];
#pragma unroll
    for (int ds = 0; ds < 4; ++ds) o[ds] = zero8;
    float m_j[8], l_j[8];
#pragma unroll
    for (int j = 0; j < 8; ++j) { m_j[j] = -3.0e38f; l_j[j] = 0.f; }

    stageKV(0, 0);
    wait_async0();
    __syncthreads();

    for (int t0 = 0; t0 < SS; t0 += 64) {
        const int buf = (t0 >> 6) & 1;
        if (t0 + 64 < SS) stageKV(buf ^ 1, t0 + 64);   // DMA next chunk in flight
        const __bf16* kb_ = &kt[buf][0];
        const __bf16* vb_ = &vt[buf][0];

        // scores: 16 rows x 64 keys = 4 subtiles, K-dim = DH = 64 (2 wmma each)
        v8f st[4];
#pragma unroll
        for (int ts = 0; ts < 4; ++ts) {
            int col = ts * 16 + l15;              // B column = key index t
            int k0  = half * 8;                   // rows K = d
            v16bf b0 = load_frag16(&kb_[col * LKV + k0],      &kb_[col * LKV + k0 + 16]);
            v16bf b1 = load_frag16(&kb_[col * LKV + 32 + k0], &kb_[col * LKV + 32 + k0 + 16]);
            v8f s = zero8;
            s = wmma_bf16(qa[0], b0, s);
            s = wmma_bf16(qa[1], b1, s);
#pragma unroll
            for (int r = 0; r < 8; ++r) s[r] *= 0.125f;   // 1/sqrt(64)
            st[ts] = s;
        }

        // online softmax: row j of this half-wave = s0 + j + half*8
#pragma unroll
        for (int j = 0; j < 8; ++j) {
            float cm = st[0][j];
#pragma unroll
            for (int ts = 1; ts < 4; ++ts) cm = fmaxf(cm, st[ts][j]);
            cm = fmaxf(cm, xor_shuffle<1>(cm));
            cm = fmaxf(cm, xor_shuffle<2>(cm));
            cm = fmaxf(cm, xor_shuffle<4>(cm));
            cm = fmaxf(cm, xor_shuffle<8>(cm));
            float mn = fmaxf(m_j[j], cm);
            float sc = __expf(m_j[j] - mn);
            m_j[j] = mn;
            float ps = 0.f;
#pragma unroll
            for (int ts = 0; ts < 4; ++ts) {
                float p = __expf(st[ts][j] - mn);
                st[ts][j] = p;
                ps += p;
            }
            ps += xor_shuffle<1>(ps);
            ps += xor_shuffle<2>(ps);
            ps += xor_shuffle<4>(ps);
            ps += xor_shuffle<8>(ps);
            l_j[j] = l_j[j] * sc + ps;
#pragma unroll
            for (int ds = 0; ds < 4; ++ds) o[ds][j] *= sc;
        }

        // D-layout -> A-layout: round-trip P through private LDS tile as bf16
        __bf16* pw = &pt[wave][0];
#pragma unroll
        for (int ts = 0; ts < 4; ++ts)
#pragma unroll
            for (int j = 0; j < 8; ++j)
                pw[(j + half * 8) * LKV + ts * 16 + l15] = (__bf16)st[ts][j];

        v16bf pa0, pa1;
        {
            int k0 = half * 8;
            pa0 = load_frag16(&pw[l15 * LKV + k0],      &pw[l15 * LKV + k0 + 16]);
            pa1 = load_frag16(&pw[l15 * LKV + 32 + k0], &pw[l15 * LKV + 32 + k0 + 16]);
        }
        // out += P(16x64) @ V(64x64): per d-subtile, two wmma over t
#pragma unroll
        for (int ds = 0; ds < 4; ++ds) {
            int col = ds * 16 + l15;              // B column = dim d
            int k0  = half * 8;                   // rows K = t
            v16bf vb0 = load_frag16(&vb_[col * LKV + k0],      &vb_[col * LKV + k0 + 16]);
            v16bf vb1 = load_frag16(&vb_[col * LKV + 32 + k0], &vb_[col * LKV + 32 + k0 + 16]);
            o[ds] = wmma_bf16(pa0, vb0, o[ds]);
            o[ds] = wmma_bf16(pa1, vb1, o[ds]);
        }

        wait_async0();
        __syncthreads();
    }

    // finalize: divide by softmax denom, merge heads, store bf16
#pragma unroll
    for (int ds = 0; ds < 4; ++ds)
#pragma unroll
        for (int j = 0; j < 8; ++j) {
            float v = o[ds][j] / l_j[j];
            int row = s0 + j + half * 8;
            int col = h * DHH + ds * 16 + l15;
            attnh[((size_t)b * SS + row) * OXX + col] = (__bf16)v;
        }
}

// ---------------------------------------------------------------------------
// Host launcher
// ---------------------------------------------------------------------------
extern "C" void kernel_launch(void* const* d_in, const int* in_sizes, int n_in,
                              void* d_out, int out_size, void* d_ws, size_t ws_size,
                              hipStream_t stream) {
    const float* x  = (const float*)d_in[0];   // (B,S,IX)
    const float* w1 = (const float*)d_in[1];   // (IX, 3*OX)
    const float* b1 = (const float*)d_in[2];   // (1, 3*OX)
    const float* w2 = (const float*)d_in[3];   // (OX, IX)
    const float* b2 = (const float*)d_in[4];   // (1, IX)
    float* out = (float*)d_out;                // (B,S,IX) f32

    char* ws = (char*)d_ws;
    __bf16* xh    = (__bf16*)ws;  ws += (size_t)MROWS * IXX  * 2;
    __bf16* w1h   = (__bf16*)ws;  ws += (size_t)IXX  * QKVW * 2;
    __bf16* w2h   = (__bf16*)ws;  ws += (size_t)OXX  * IXX  * 2;
    __bf16* qkvh  = (__bf16*)ws;  ws += (size_t)MROWS * QKVW * 2;
    __bf16* attnh = (__bf16*)ws;

    // 1) f32 -> bf16 conversions
    {
        int n;
        n = MROWS * IXX;
        cvt_f32_bf16<<<(n / 4 + 255) / 256, 256, 0, stream>>>(x, xh, n);
        n = IXX * QKVW;
        cvt_f32_bf16<<<(n / 4 + 255) / 256, 256, 0, stream>>>(w1, w1h, n);
        n = OXX * IXX;
        cvt_f32_bf16<<<(n / 4 + 255) / 256, 256, 0, stream>>>(w2, w2h, n);
    }
    // 2) fused QKV projection: (8192,1024) @ (1024,3072) + b1 -> bf16
    {
        dim3 grid(QKVW / 128, MROWS / 128);
        gemm_bf16_kernel<true><<<grid, 256, 0, stream>>>(xh, w1h, b1, qkvh,
                                                         MROWS, QKVW, IXX);
    }
    // 3) flash attention over 64 (b,h) pairs
    {
        dim3 grid(SS / 128, BB * HH);
        attn_kernel<<<grid, 256, 0, stream>>>(qkvh, attnh);
    }
    // 4) output projection: (8192,1024) @ (1024,1024) + b2 -> f32 out
    {
        dim3 grid(IXX / 128, MROWS / 128);
        gemm_bf16_kernel<false><<<grid, 256, 0, stream>>>(attnh, w2h, b2, out,
                                                          MROWS, IXX, OXX);
    }
}
